// PPAM_DSAformer_21062519619705
// MI455X (gfx1250) — matrix operand order repaired and checked by reading, not compiled, not run on hardware
//
#include <hip/hip_runtime.h>
#include <hip/hip_bf16.h>
#include <math.h>

// ---------------------------------------------------------------------------
// CDNA5 (gfx1250) axial-attention pipeline.
//   - All large GEMMs on v_wmma_f32_16x16x32_bf16 (bf16 in, fp32 accum).
//   - A tile stored in LDS in WMMA fragment order -> 2x ds_load_b128 / frag.
//   - Software pipelined: global gathers for k-step ks+1 issue between the
//     staging barrier and the WMMA block of k-step ks.
//   - fp32->bf16 via native v_cvt_pk_bf16_f32 (builtin-guarded).
//   - Attention core (d<=13) is VALU in LDS.
// ---------------------------------------------------------------------------

typedef __attribute__((ext_vector_type(16))) __bf16 bf16x16;
typedef __attribute__((ext_vector_type(2)))  __bf16 bf16x2;
typedef __attribute__((ext_vector_type(8)))  float  f32x8;

#define BDIM   512
#define CCH    384
#define HDIM   5
#define WDIM   13
#define SPA    65              // H*W
#define NCOLS  33280           // B*H*W = B'*d for every pass
#define PLANE  24960           // C*SPA (per-batch stride, canonical NCHW)
#define EPS    1e-5f

// Pack two fp32 into a dword of two bf16 using the hardware packed convert
// when available; otherwise clang's native fptrunc lowering.
static __device__ __forceinline__ unsigned pack2bf(float lo, float hi) {
#if __has_builtin(__builtin_amdgcn_cvt_pk_bf16_f32)
    bf16x2 v = __builtin_amdgcn_cvt_pk_bf16_f32(lo, hi);
    return __builtin_bit_cast(unsigned, v);
#else
    bf16x2 v;
    v[0] = (__bf16)lo;
    v[1] = (__bf16)hi;
    return __builtin_bit_cast(unsigned, v);
#endif
}

struct U8 { uint4 lo, hi; };
static __device__ __forceinline__ bf16x16 ldsFrag(const unsigned* p) {
    U8 u;
    u.lo = *(const uint4*)p;
    u.hi = *(const uint4*)(p + 4);
    return __builtin_bit_cast(bf16x16, u);
}

// Gather one element of the (K x N) RHS matrix from canonical NCHW buffers.
// mode 0: H-axis pass   (n = (b*13+w)*5 + h)
// mode 1: W-axis pass   (n = (b*5+h)*13 + w)  -> contiguous within 65-block
// mode 2: concat(bufC, bufD) for the final projection (K = 768)
static __device__ __forceinline__ float gatherB(const float* __restrict__ s0,
                                                const float* __restrict__ s1,
                                                int mode, int kk, int n) {
    unsigned un  = (unsigned)n;
    unsigned b   = un / 65u;
    unsigned rem = un % 65u;
    if (mode == 0) {
        unsigned w = rem / 5u, h = rem % 5u;
        return s0[(size_t)b * PLANE + (size_t)kk * SPA + h * 13u + w];
    } else if (mode == 1) {
        return s0[(size_t)b * PLANE + (size_t)kk * SPA + rem];
    } else {
        const float* p = (kk < CCH) ? s0 : s1;
        int k2 = (kk < CCH) ? kk : kk - CCH;
        return p[(size_t)b * PLANE + (size_t)k2 * SPA + rem];
    }
}

// ---------------------------------------------------------------------------
// Tiled WMMA GEMM:  C[M,N] = A[M,K] * X[K,N]   (N = 33280 always)
//   gridDim = (N/128, M/128), 256 threads (8 waves), tile 128x128x32.
//   epiMode 0: BN then store transposed  out[n*M + m]  (QKV ws, b128 stores)
//   epiMode 1: BN + ReLU then scatter to canonical    (final output)
// ---------------------------------------------------------------------------
__global__ __launch_bounds__(256)
void axial_gemm_wmma(const float* __restrict__ A,
                     const float* __restrict__ Bsrc0,
                     const float* __restrict__ Bsrc1,
                     const float* __restrict__ bn,
                     float* __restrict__ out,
                     int M, int K, int gatherMode, int epiMode) {
    // A tile, fragment-ordered: [mtile 0..7][lane 0..31][8 x u32 = 16 bf16]
    __shared__ unsigned sA32[8 * 32 * 8];     // 8 KB
    // B tile, fragment-ordered: [ntile 0..7][lane 0..31][8 x u32 = 16 bf16] + unused tail
    __shared__ unsigned sB32[32 * 68];        // 8.5 KB

    const int t     = threadIdx.x;
    const int lane  = t & 31;
    const int wave  = t >> 5;
    const int waveM = wave >> 1;              // 0..3 -> 2 M-subtiles each
    const int waveN = wave & 1;               // 0..1 -> 4 N-subtiles each
    const int bM    = blockIdx.y * 128;
    const int bN    = blockIdx.x * 128;

    float2 aReg[8];                           // staged A pairs (k, k+1)
    float2 bReg[8];                           // staged B pairs (k, k+1)
    f32x8  acc[2][4] = {};

    const int Ksteps = K >> 5;

    // ---- prologue: fetch k-step 0 into registers ----
#pragma unroll
    for (int e = 0; e < 8; ++e) {
        int idx = e * 256 + t;
        int m = idx >> 4, kp = idx & 15;
        const float* p = &A[(size_t)(bM + m) * K + (kp << 1)];
        aReg[e] = make_float2(p[0], p[1]);
        int n = idx & 127, kq = idx >> 7;
        bReg[e].x = gatherB(Bsrc0, Bsrc1, gatherMode, kq << 1, bN + n);
        bReg[e].y = gatherB(Bsrc0, Bsrc1, gatherMode, (kq << 1) + 1, bN + n);
    }

    for (int ks = 0; ks < Ksteps; ++ks) {
        // ---- commit staged registers to LDS (packed bf16 pairs) ----
#pragma unroll
        for (int e = 0; e < 8; ++e) {
            int idx = e * 256 + t;
            {   // A: swizzle into fragment order
                int m = idx >> 4, kk0 = (idx & 15) << 1;
                int half = (kk0 >> 3) & 1;
                int r = ((kk0 & 16) >> 2) | ((kk0 & 7) >> 1);
                sA32[(((m >> 4) << 5) | ((m & 15) | (half << 4))) * 8 + r] =
                    pack2bf(aReg[e].x, aReg[e].y);
            }
            {   // B: swizzle into fragment order
                int n = idx & 127, kk0 = (idx >> 7) << 1;
                int half = (kk0 >> 3) & 1;
                int r = ((kk0 & 16) >> 2) | ((kk0 & 7) >> 1);
                sB32[(((n >> 4) << 5) | ((n & 15) | (half << 4))) * 8 + r] =
                    pack2bf(bReg[e].x, bReg[e].y);
            }
        }
        __syncthreads();

        // ---- prefetch k-step ks+1 into registers (overlaps the WMMAs) ----
        if (ks + 1 < Ksteps) {
            const int k0 = (ks + 1) << 5;
#pragma unroll
            for (int e = 0; e < 8; ++e) {
                int idx = e * 256 + t;
                int m = idx >> 4, kp = idx & 15;
                const float* p = &A[(size_t)(bM + m) * K + k0 + (kp << 1)];
                aReg[e] = make_float2(p[0], p[1]);
                int n = idx & 127, kq = idx >> 7;
                bReg[e].x = gatherB(Bsrc0, Bsrc1, gatherMode, k0 + (kq << 1), bN + n);
                bReg[e].y = gatherB(Bsrc0, Bsrc1, gatherMode, k0 + (kq << 1) + 1, bN + n);
            }
        }

        // ---- fragments (all b128 LDS reads) + 8 WMMAs ----
        bf16x16 afr[2], bfr[4];
#pragma unroll
        for (int im = 0; im < 2; ++im)
            afr[im] = ldsFrag(&sA32[((((waveM * 2 + im) << 5) | lane) << 3)]);
#pragma unroll
        for (int in = 0; in < 4; ++in)
            bfr[in] = ldsFrag(&sB32[((((waveN * 4 + in) << 5) | lane) << 3)]);
#pragma unroll
        for (int im = 0; im < 2; ++im)
#pragma unroll
            for (int in = 0; in < 4; ++in)
                acc[im][in] = __builtin_amdgcn_wmma_f32_16x16x32_bf16(
                    false, afr[im], false, bfr[in], (short)0, acc[im][in],
                    false, false);
        __syncthreads();
    }

    // ---- epilogue: fused BN (+ReLU/scatter) ----
    const int nloc  = lane & 15;
    const int mhalf = (lane >> 4) * 8;
#pragma unroll
    for (int im = 0; im < 2; ++im) {
        const int mBase = bM + (waveM * 2 + im) * 16 + mhalf;
        float sc[8], sh[8];
#pragma unroll
        for (int r = 0; r < 8; ++r) {
            int m = mBase + r;
            float s = bn[m] * rsqrtf(bn[3 * M + m] + EPS);
            sc[r] = s;
            sh[r] = bn[M + m] - bn[2 * M + m] * s;
        }
#pragma unroll
        for (int in = 0; in < 4; ++in) {
            const int n = bN + (waveN * 4 + in) * 16 + nloc;
            if (epiMode == 0) {
                float4 lo, hi;
                lo.x = acc[im][in][0] * sc[0] + sh[0];
                lo.y = acc[im][in][1] * sc[1] + sh[1];
                lo.z = acc[im][in][2] * sc[2] + sh[2];
                lo.w = acc[im][in][3] * sc[3] + sh[3];
                hi.x = acc[im][in][4] * sc[4] + sh[4];
                hi.y = acc[im][in][5] * sc[5] + sh[5];
                hi.z = acc[im][in][6] * sc[6] + sh[6];
                hi.w = acc[im][in][7] * sc[7] + sh[7];
                float* dst = &out[(size_t)n * M + mBase];
                *(float4*)dst       = lo;        // two global_store_b128
                *(float4*)(dst + 4) = hi;
            } else {
                unsigned b = (unsigned)n / 65u, rem = (unsigned)n % 65u;
                float* dst = &out[(size_t)b * PLANE + rem];
#pragma unroll
                for (int r = 0; r < 8; ++r) {
                    float v = fmaxf(acc[im][in][r] * sc[r] + sh[r], 0.0f);
                    dst[(size_t)(mBase + r) * SPA] = v;
                }
            }
        }
    }
}

// ---------------------------------------------------------------------------
// Attention core: one workgroup per batch-line b'. Reads BN'd QKV columns
// (layout [b'*d + j][512]), computes qr/kr/dots -> bn_attn -> softmax ->
// out/kv -> bn_out -> sum, scatters y[c,dpos] back into a canonical buffer.
//   scatterMode 0: H pass (b'=b*13+w, dpos=h)   d = 5
//   scatterMode 1: W pass (b'=b*5+h,  dpos=w)   d = 13
// ---------------------------------------------------------------------------
__global__ __launch_bounds__(256)
void axial_attn(const float* __restrict__ qkv,
                const float* __restrict__ rel,
                const float* __restrict__ bn_attn,
                const float* __restrict__ bn_out,
                float* __restrict__ out,
                int d, int scatterMode) {
    __shared__ float sqkv[512 * 13];     // [j][m]
    __shared__ float srel[64 * 25];      // [row][2d-1]
    __shared__ float ssc[8 * 13 * 13];   // scores / attn
    __shared__ float sabn[2 * 24];       // bn_attn scale|shift
    __shared__ float sobn[2 * 768];      // bn_out scale|shift

    const int bp = blockIdx.x;
    const int t  = threadIdx.x;
    const int rw = 2 * d - 1;

    const float* src = qkv + (size_t)bp * d * 512;
    for (int i = t; i < 512 * d; i += 256) sqkv[i] = src[i];
    for (int i = t; i < 64 * rw; i += 256) srel[i] = rel[i];
    if (t < 24) {
        float sc = bn_attn[t] * rsqrtf(bn_attn[72 + t] + EPS);
        sabn[t]      = sc;
        sabn[24 + t] = bn_attn[24 + t] - bn_attn[48 + t] * sc;
    }
    for (int i = t; i < 768; i += 256) {
        float sc = bn_out[i] * rsqrtf(bn_out[3 * 768 + i] + EPS);
        sobn[i]       = sc;
        sobn[768 + i] = bn_out[768 + i] - bn_out[2 * 768 + i] * sc;
    }
    __syncthreads();

    // scores: qr + kr + dots, each BN'd over its 24-channel stack
    const int dd = d * d;
    for (int idx = t; idx < 8 * dd; idx += 256) {
        int head = idx / dd, r2 = idx % dd;
        int dp = r2 / d, j = r2 % d;
        int roff = dp - j + d - 1;
        float qr = 0.f, kr = 0.f, dot = 0.f;
#pragma unroll
        for (int i = 0; i < 8; ++i) {
            float qv = sqkv[dp * 512 + (i * 8 + head)];
            float kd = sqkv[dp * 512 + ((8 + i) * 8 + head)];
            float kj = sqkv[j  * 512 + ((8 + i) * 8 + head)];
            qr  += qv * srel[i * rw + roff];
            kr  += kd * srel[(8 + i) * rw + roff];
            dot += qv * kj;
        }
        int c0 = head * 3;
        ssc[idx] = (qr  * sabn[c0]     + sabn[24 + c0])
                 + (kr  * sabn[c0 + 1] + sabn[24 + c0 + 1])
                 + (dot * sabn[c0 + 2] + sabn[24 + c0 + 2]);
    }
    __syncthreads();

    // softmax over j for each (head, dpos)
    for (int r = t; r < 8 * d; r += 256) {
        float* row = &ssc[r * d];
        float mx = row[0];
        for (int j = 1; j < d; ++j) mx = fmaxf(mx, row[j]);
        float sum = 0.f;
        for (int j = 0; j < d; ++j) { float e = __expf(row[j] - mx); row[j] = e; sum += e; }
        float inv = 1.f / sum;
        for (int j = 0; j < d; ++j) row[j] *= inv;
    }
    __syncthreads();

    // out (attn @ v) + kv (attn @ r_v), bn_out, sum of halves, scatter
    for (int idx = t; idx < 384 * d; idx += 256) {
        int c = idx / d, dp = idx % d;
        int head = c / 48, iv = c % 48;
        const float* arow = &ssc[(head * d + dp) * d];
        float o = 0.f, kvv = 0.f;
        for (int j = 0; j < d; ++j) {
            float a = arow[j];
            o   += a * sqkv[j * 512 + ((16 + iv) * 8 + head)];
            kvv += a * srel[(16 + iv) * rw + (dp - j + d - 1)];
        }
        float y = (kvv * sobn[c]       + sobn[768 + c])
                + (o   * sobn[384 + c] + sobn[768 + 384 + c]);
        int b, h, w;
        if (scatterMode == 0) { b = bp / 13; w = bp % 13; h = dp; }
        else                  { b = bp / 5;  h = bp % 5;  w = dp; }
        out[(size_t)b * PLANE + (size_t)c * SPA + h * 13 + w] = y;
    }
}

// ---------------------------------------------------------------------------
extern "C" void kernel_launch(void* const* d_in, const int* in_sizes, int n_in,
                              void* d_out, int out_size, void* d_ws, size_t ws_size,
                              hipStream_t stream) {
    (void)in_sizes; (void)n_in; (void)out_size; (void)ws_size;
    const float* x       = (const float*)d_in[0];
    const float* WqkvH   = (const float*)d_in[1];
    const float* bnqkvH  = (const float*)d_in[2];
    const float* relH    = (const float*)d_in[3];
    const float* bnattnH = (const float*)d_in[4];
    const float* bnoutH  = (const float*)d_in[5];
    const float* WqkvV   = (const float*)d_in[6];
    const float* bnqkvV  = (const float*)d_in[7];
    const float* relV    = (const float*)d_in[8];
    const float* bnattnV = (const float*)d_in[9];
    const float* bnoutV  = (const float*)d_in[10];
    const float* Wout    = (const float*)d_in[11];
    const float* bnfinal = (const float*)d_in[12];
    float* out = (float*)d_out;

    // workspace: QKV scratch + 4 canonical NCHW buffers
    const size_t qkvFloats = (size_t)NCOLS * 512;          // 68 MB
    const size_t bufFloats = (size_t)BDIM * CCH * SPA;     // 51 MB each
    float* qkvW = (float*)d_ws;
    float* bufA = qkvW + qkvFloats;
    float* bufB = bufA + bufFloats;
    float* bufC = bufB + bufFloats;
    float* bufD = bufC + bufFloats;

    const dim3 gQKV(NCOLS / 128, 512 / 128);   // (260, 4)
    const dim3 gFin(NCOLS / 128, 384 / 128);   // (260, 3)

    // pass 1: H-axis attention on x -> bufA
    axial_gemm_wmma<<<gQKV, 256, 0, stream>>>(WqkvH, x, nullptr, bnqkvH, qkvW, 512, 384, 0, 0);
    axial_attn<<<BDIM * WDIM, 256, 0, stream>>>(qkvW, relH, bnattnH, bnoutH, bufA, HDIM, 0);
    // pass 2: W-axis attention on x -> bufB
    axial_gemm_wmma<<<gQKV, 256, 0, stream>>>(WqkvV, x, nullptr, bnqkvV, qkvW, 512, 384, 1, 0);
    axial_attn<<<BDIM * HDIM, 256, 0, stream>>>(qkvW, relV, bnattnV, bnoutV, bufB, WDIM, 1);
    // pass 3: W-axis attention on bufA (o_HV) -> bufC
    axial_gemm_wmma<<<gQKV, 256, 0, stream>>>(WqkvV, bufA, nullptr, bnqkvV, qkvW, 512, 384, 1, 0);
    axial_attn<<<BDIM * HDIM, 256, 0, stream>>>(qkvW, relV, bnattnV, bnoutV, bufC, WDIM, 1);
    // pass 4: H-axis attention on bufB (o_VH) -> bufD
    axial_gemm_wmma<<<gQKV, 256, 0, stream>>>(WqkvH, bufB, nullptr, bnqkvH, qkvW, 512, 384, 0, 0);
    axial_attn<<<BDIM * WDIM, 256, 0, stream>>>(qkvW, relH, bnattnH, bnoutH, bufD, HDIM, 0);
    // final: W_out @ concat(bufC, bufD) + BN + ReLU -> out (canonical NCHW)
    axial_gemm_wmma<<<gFin, 256, 0, stream>>>(Wout, bufC, bufD, bnfinal, out, 384, 768, 2, 1);
}
